// PointNet2Segmentation_3143916061383
// MI455X (gfx1250) — matrix operand order repaired and checked by reading, not compiled, not run on hardware
//
#include <hip/hip_runtime.h>
#include <hip/hip_bf16.h>

typedef __attribute__((ext_vector_type(16))) _Float16 v16h;
typedef __attribute__((ext_vector_type(8)))  _Float16 v8h;
typedef __attribute__((ext_vector_type(8)))  float    v8f;

#define GSL(i, total) for (long i = (long)blockIdx.x * blockDim.x + threadIdx.x; \
                           i < (total); i += (long)gridDim.x * blockDim.x)

// ---------------------------------------------------------------------------
// Layout helpers / misc kernels
// ---------------------------------------------------------------------------

// x: (B, Cx, N) -> out: (B, N, Cx)
__global__ void transpose_in_kernel(const float* __restrict__ x, float* __restrict__ out,
                                    int B, int Cx, int N, long total) {
    GSL(i, total) {
        int c = (int)(i % Cx);
        long r = i / Cx;
        int n = (int)(r % N);
        int b = (int)(r / N);
        out[i] = x[((size_t)b * Cx + c) * N + n];
    }
}

// logits: (B, N, C) -> out: (B, C, N)
__global__ void transpose_out_kernel(const float* __restrict__ in, float* __restrict__ out,
                                     int B, int N, int C, long total) {
    GSL(i, total) {
        int n = (int)(i % N);
        long r = i / N;
        int c = (int)(r % C);
        int b = (int)(r / C);
        out[i] = in[((size_t)b * N + n) * C + c];
    }
}

// out(q, Ca+Cb) = concat(a(q, Ca), b(q, Cb))
__global__ void concat2_kernel(const float* __restrict__ a, const float* __restrict__ b,
                               float* __restrict__ out, long BS, int Ca, int Cb, long total) {
    int C = Ca + Cb;
    GSL(i, total) {
        int c = (int)(i % C);
        long q = i / C;
        out[i] = (c < Ca) ? a[q * Ca + c] : b[q * Cb + (c - Ca)];
    }
}

// ---------------------------------------------------------------------------
// Farthest point sampling: one block per batch, dist array resident in LDS
// ---------------------------------------------------------------------------
__global__ __launch_bounds__(256)
void fps_kernel(const float* __restrict__ xyz, int* __restrict__ idx, int N, int npoint) {
    const int b = blockIdx.x;
    const float* p = xyz + (size_t)b * N * 3;
    __shared__ float dist[4096];
    __shared__ float rv[256];
    __shared__ int   ri[256];
    __shared__ int   sfar;
    for (int j = threadIdx.x; j < N; j += 256) dist[j] = 1e10f;
    if (threadIdx.x == 0) sfar = 0;
    __syncthreads();
    for (int s = 0; s < npoint; ++s) {
        int far = sfar;
        if (threadIdx.x == 0) idx[b * npoint + s] = far;
        float fx = p[far * 3 + 0], fy = p[far * 3 + 1], fz = p[far * 3 + 2];
        float bestv = -1.0f; int besti = 0;
        for (int j = threadIdx.x; j < N; j += 256) {
            float dx = p[j * 3 + 0] - fx;
            float dy = p[j * 3 + 1] - fy;
            float dz = p[j * 3 + 2] - fz;
            float d = dx * dx + dy * dy + dz * dz;
            float dj = dist[j];
            if (d < dj) { dj = d; dist[j] = d; }
            if (dj > bestv) { bestv = dj; besti = j; }
        }
        rv[threadIdx.x] = bestv; ri[threadIdx.x] = besti;
        __syncthreads();
        for (int off = 128; off > 0; off >>= 1) {
            if (threadIdx.x < off) {
                float v2 = rv[threadIdx.x + off]; int i2 = ri[threadIdx.x + off];
                if (v2 > rv[threadIdx.x] || (v2 == rv[threadIdx.x] && i2 < ri[threadIdx.x])) {
                    rv[threadIdx.x] = v2; ri[threadIdx.x] = i2;
                }
            }
            __syncthreads();
        }
        if (threadIdx.x == 0) sfar = ri[0];
        __syncthreads();
    }
}

__global__ void gather_new_xyz_kernel(const float* __restrict__ xyz, const int* __restrict__ fi,
                                      float* __restrict__ nx, int N, int S, long total) {
    GSL(q, total) {
        int b = (int)(q / S);
        int pidx = fi[q];
        const float* src = xyz + ((size_t)b * N + pidx) * 3;
        nx[q * 3 + 0] = src[0];
        nx[q * 3 + 1] = src[1];
        nx[q * 3 + 2] = src[2];
    }
}

// ---------------------------------------------------------------------------
// Ball query: thread per query, first nsample neighbors in index order
// ---------------------------------------------------------------------------
__global__ void ball_query_kernel(const float* __restrict__ xyz, const float* __restrict__ nxyz,
                                  int* __restrict__ idx, int N, int S, float r2, int ns, long total) {
    GSL(q, total) {
        int b = (int)(q / S);
        const float* p = xyz + (size_t)b * N * 3;
        float cx = nxyz[q * 3 + 0], cy = nxyz[q * 3 + 1], cz = nxyz[q * 3 + 2];
        int* o = idx + q * ns;
        int cnt = 0;
        for (int j = 0; j < N && cnt < ns; ++j) {
            float dx = p[j * 3 + 0] - cx;
            float dy = p[j * 3 + 1] - cy;
            float dz = p[j * 3 + 2] - cz;
            float d = dx * dx + dy * dy + dz * dz;
            if (d <= r2) { o[cnt++] = j; }
        }
        if (cnt == 0) { o[0] = 0; cnt = 1; }
        int f0 = o[0];
        for (int t = cnt; t < ns; ++t) o[t] = f0;
    }
}

// out (B,S,ns,3+Cp): first 3 channels = gathered xyz - center; rest = gathered pts
__global__ void group_points_kernel(const float* __restrict__ xyz, const float* __restrict__ nxyz,
                                    const float* __restrict__ pts, const int* __restrict__ idx,
                                    float* __restrict__ out, int N, int S, int ns, int Cp, long total) {
    int C = 3 + Cp;
    GSL(i, total) {
        int c = (int)(i % C);
        long r = i / C;          // row = (b*S+s)*ns + j
        long q = r / ns;         // (b*S+s)
        int b = (int)(q / S);
        int pidx = idx[r];
        float v;
        if (c < 3) v = xyz[((size_t)b * N + pidx) * 3 + c] - nxyz[q * 3 + c];
        else       v = pts[((size_t)b * N + pidx) * Cp + (c - 3)];
        out[i] = v;
    }
}

// ---------------------------------------------------------------------------
// GEMM path: f32 -> K-padded f16 activations, fragment-packed f16 weights,
// wave32 WMMA 16x16x32 f16 with f32 accumulation.
// ---------------------------------------------------------------------------
__global__ void cvt_pad_f16_kernel(const float* __restrict__ in, _Float16* __restrict__ out,
                                   long M, int C, int Kp) {
    long total = M * Kp;
    GSL(i, total) {
        long m = i / Kp;
        int k = (int)(i % Kp);
        out[i] = (k < C) ? (_Float16)in[(size_t)m * C + k] : (_Float16)0.0f;
    }
}

// Pack W(Co,Ci) into B-fragment-ready order: [ntile][kstep][lane][16 halves],
// lane<16: K = kk*32 + (i<8 ? i : 16+i-8), col = nt*16+lane; lane>=16: K = kk*32 + (i<8 ? 8+i : 24+i-8), col = nt*16+(lane-16)
// ntile dimension is rounded up to a multiple of 4 (zero-filled) so the GEMM
// can blindly process 4 N-tiles per wave with full EXEC.
__global__ void pack_weights_kernel(const float* __restrict__ W, _Float16* __restrict__ Wp,
                                    int Co, int Ci, int Kp, long total) {
    int ksteps = Kp >> 5;
    GSL(i, total) {
        int e = (int)(i & 15);
        int lane = (int)((i >> 4) & 31);
        long r = i >> 9;                 // nt*ksteps + kk
        int kk = (int)(r % ksteps);
        int nt = (int)(r / ksteps);
        int col = nt * 16 + (lane & 15);
        int k = kk * 32 + ((e >> 3) << 4) + ((lane >> 4) << 3) + (e & 7);
        Wp[i] = (col < Co && k < Ci) ? (_Float16)W[(size_t)col * Ci + k] : (_Float16)0.0f;
    }
}

__device__ __forceinline__ void store_tile(float* __restrict__ out, const float* __restrict__ bias,
                                           v8f c, int rbase, int col, int Co) {
    if (col < Co) {
        float bv = bias ? bias[col] : 0.0f;
#pragma unroll
        for (int r = 0; r < 8; ++r)
            out[(size_t)(rbase + r) * Co + col] = c[r] + bv;
    }
}

// out(M,Co) = A(M,Kp) * Wp + bias.
// 4 waves/block; each wave: 1 M-tile x 4 N-tiles (A fragment reused 4x per K-step).
__global__ __launch_bounds__(128)
void gemm_wmma_kernel(const _Float16* __restrict__ A, const _Float16* __restrict__ Wp,
                      const float* __restrict__ bias, float* __restrict__ out,
                      int M, int Kp, int Co) {
    const int lane = threadIdx.x & 31;
    const int wave = threadIdx.x >> 5;
    const int mtile = blockIdx.x * 4 + wave;
    const int ntile0 = blockIdx.y * 4;
    const int ksteps = Kp >> 5;
    const int row0 = mtile * 16;
    const int halfsel = lane >> 4;     // 0 or 1
    const int lidx = lane & 15;

    const _Float16* aptr = A + (size_t)(row0 + lidx) * Kp + halfsel * 8;
    const size_t tstride = (size_t)ksteps * 512;   // halves per packed N-tile
    const _Float16* bptr = Wp + (size_t)ntile0 * tstride + (size_t)lane * 16;

    v8f c0 = {}, c1 = {}, c2 = {}, c3 = {};
    for (int kk = 0; kk < ksteps; ++kk) {
        v8h alo = *(const v8h*)(aptr);
        v8h ahi = *(const v8h*)(aptr + 16);
        v16h af;
#pragma unroll
        for (int i = 0; i < 8; ++i) { af[i] = alo[i]; af[i + 8] = ahi[i]; }
        v16h b0 = *(const v16h*)(bptr);
        v16h b1 = *(const v16h*)(bptr + tstride);
        v16h b2 = *(const v16h*)(bptr + 2 * tstride);
        v16h b3 = *(const v16h*)(bptr + 3 * tstride);
        c0 = __builtin_amdgcn_wmma_f32_16x16x32_f16(false, af, false, b0, (short)0, c0, false, false);
        c1 = __builtin_amdgcn_wmma_f32_16x16x32_f16(false, af, false, b1, (short)0, c1, false, false);
        c2 = __builtin_amdgcn_wmma_f32_16x16x32_f16(false, af, false, b2, (short)0, c2, false, false);
        c3 = __builtin_amdgcn_wmma_f32_16x16x32_f16(false, af, false, b3, (short)0, c3, false, false);
        aptr += 32;
        bptr += 512;
        __builtin_prefetch(aptr, 0, 3);   // speculative global_prefetch_b8, no guard needed
    }

    const int rbase = row0 + halfsel * 8;
    store_tile(out, bias, c0, rbase, (ntile0 + 0) * 16 + lidx, Co);
    store_tile(out, bias, c1, rbase, (ntile0 + 1) * 16 + lidx, Co);
    store_tile(out, bias, c2, rbase, (ntile0 + 2) * 16 + lidx, Co);
    store_tile(out, bias, c3, rbase, (ntile0 + 3) * 16 + lidx, Co);
}

// ---------------------------------------------------------------------------
// BatchNorm stats (one block per channel) + normalize + ReLU
// ---------------------------------------------------------------------------
__global__ __launch_bounds__(256)
void bn_stats_kernel(const float* __restrict__ h, float* __restrict__ mean,
                     float* __restrict__ var, long M, int C) {
    const int c = blockIdx.x;
    float s = 0.0f, s2 = 0.0f;
    for (long m = threadIdx.x; m < M; m += 256) {
        float v = h[(size_t)m * C + c];
        s += v; s2 += v * v;
    }
    __shared__ float rs[256], rs2[256];
    rs[threadIdx.x] = s; rs2[threadIdx.x] = s2;
    __syncthreads();
    for (int off = 128; off > 0; off >>= 1) {
        if (threadIdx.x < off) {
            rs[threadIdx.x]  += rs[threadIdx.x + off];
            rs2[threadIdx.x] += rs2[threadIdx.x + off];
        }
        __syncthreads();
    }
    if (threadIdx.x == 0) {
        float mu = rs[0] / (float)M;
        mean[c] = mu;
        var[c] = rs2[0] / (float)M - mu * mu;
    }
}

__global__ void bn_relu_kernel(const float* __restrict__ in, float* __restrict__ out,
                               const float* __restrict__ mean, const float* __restrict__ var,
                               long total, int C) {
    GSL(i, total) {
        int c = (int)(i % C);
        float v = (in[i] - mean[c]) * rsqrtf(var[c] + 1e-5f);
        out[i] = v > 0.0f ? v : 0.0f;
    }
}

// in (BS, ns, C) -> out (BS, C): max over ns
__global__ void maxpool_kernel(const float* __restrict__ in, float* __restrict__ out,
                               long BS, int ns, int C, long total) {
    GSL(i, total) {
        long q = i / C;
        int c = (int)(i % C);
        float m = -3.4e38f;
        for (int j = 0; j < ns; ++j)
            m = fmaxf(m, in[((size_t)q * ns + j) * C + c]);
        out[i] = m;
    }
}

// ---------------------------------------------------------------------------
// 3-NN + inverse-distance interpolation for FP stages
// ---------------------------------------------------------------------------
__global__ void knn3_kernel(const float* __restrict__ xyz1, const float* __restrict__ xyz2,
                            int* __restrict__ i3, float* __restrict__ w,
                            int Nq, int S, long total) {
    GSL(q, total) {
        int b = (int)(q / Nq);
        const float* pt = xyz1 + q * 3;
        const float* p2 = xyz2 + (size_t)b * S * 3;
        float px = pt[0], py = pt[1], pz = pt[2];
        float d0 = 1e30f, d1 = 1e30f, d2 = 1e30f;
        int j0 = 0, j1 = 0, j2 = 0;
        for (int j = 0; j < S; ++j) {
            float dx = p2[j * 3 + 0] - px;
            float dy = p2[j * 3 + 1] - py;
            float dz = p2[j * 3 + 2] - pz;
            float d = dx * dx + dy * dy + dz * dz;
            if (d < d0)      { d2 = d1; j2 = j1; d1 = d0; j1 = j0; d0 = d; j0 = j; }
            else if (d < d1) { d2 = d1; j2 = j1; d1 = d;  j1 = j; }
            else if (d < d2) { d2 = d;  j2 = j; }
        }
        float w0 = 1.0f / (d0 + 1e-8f);
        float w1 = 1.0f / (d1 + 1e-8f);
        float w2 = 1.0f / (d2 + 1e-8f);
        float ws = w0 + w1 + w2;
        i3[q * 3 + 0] = j0; i3[q * 3 + 1] = j1; i3[q * 3 + 2] = j2;
        w[q * 3 + 0] = w0 / ws; w[q * 3 + 1] = w1 / ws; w[q * 3 + 2] = w2 / ws;
    }
}

// out (B,Nq,C1+C2): concat(points1, interpolated points2)
__global__ void interp_concat_kernel(const float* __restrict__ pts1, const float* __restrict__ pts2,
                                     const int* __restrict__ i3, const float* __restrict__ w,
                                     float* __restrict__ out, long total,
                                     int C1, int C2, int S, int Nq) {
    int C = C1 + C2;
    GSL(i, total) {
        int c = (int)(i % C);
        long q = i / C;
        int b = (int)(q / Nq);
        float v;
        if (c < C1) {
            v = pts1[q * C1 + c];
        } else {
            int c2 = c - C1;
            if (S == 1) {
                v = pts2[(size_t)b * C2 + c2];
            } else {
                const int* ii = i3 + q * 3;
                const float* ww = w + q * 3;
                const float* base = pts2 + (size_t)b * S * C2;
                v = ww[0] * base[(size_t)ii[0] * C2 + c2]
                  + ww[1] * base[(size_t)ii[1] * C2 + c2]
                  + ww[2] * base[(size_t)ii[2] * C2 + c2];
            }
        }
        out[i] = v;
    }
}

// ---------------------------------------------------------------------------
// Host orchestration
// ---------------------------------------------------------------------------
static inline unsigned gblocks(long total) {
    long b = (total + 255) / 256;
    if (b < 1) b = 1;
    if (b > 65535) b = 65535;
    return (unsigned)b;
}

extern "C" void kernel_launch(void* const* d_in, const int* in_sizes, int n_in,
                              void* d_out, int out_size, void* d_ws, size_t ws_size,
                              hipStream_t stream) {
    (void)in_sizes; (void)n_in; (void)out_size; (void)ws_size;
    const int B = 16, N = 4096;
    const float* x = (const float*)d_in[0];

    struct Layer { const float* W; const float* bias; int Ci; int Co; };
    static const int dims[23][2] = {
        {3,32},{32,32},{32,64},            // sa1
        {67,64},{64,64},{64,128},          // sa2
        {131,128},{128,128},{128,256},     // sa3
        {259,256},{256,512},{512,1024},    // sa4
        {1280,256},{256,256},              // fp4
        {384,256},{256,256},               // fp3
        {320,256},{256,128},               // fp2
        {131,128},{128,128},{128,128},     // fp1
        {128,128},{128,13}                 // head
    };
    Layer Ls[23];
    for (int l = 0; l < 23; ++l) {
        Ls[l].W = (const float*)d_in[1 + 2 * l];
        Ls[l].bias = (const float*)d_in[2 + 2 * l];
        Ls[l].Ci = dims[l][0]; Ls[l].Co = dims[l][1];
    }

    size_t off = 0;
    auto alloc = [&](size_t nbytes) -> void* {
        void* p = (char*)d_ws + off;
        off += (nbytes + 255) & ~(size_t)255;
        return p;
    };
    float*    xyzT  = (float*)alloc((size_t)B * N * 3 * 4);
    float*    fbuf  = (float*)alloc((size_t)34000000 * 4);    // f32 activations (M x C)
    _Float16* a16   = (_Float16*)alloc((size_t)17000000 * 2); // K-padded f16 activations
    _Float16* wpk   = (_Float16*)alloc((size_t)700000 * 2);   // fragment-packed weights
    float*    stats = (float*)alloc(2048 * 4);                // mean | var
    int*      fpsb  = (int*)alloc((size_t)B * 1024 * 4);
    int*      idxb  = (int*)alloc((size_t)B * 1024 * 32 * 4);
    int*      i3b   = (int*)alloc((size_t)B * N * 3 * 4);
    float*    wtb   = (float*)alloc((size_t)B * N * 3 * 4);
    float*    l1x   = (float*)alloc((size_t)B * 1024 * 3 * 4);
    float*    l2x   = (float*)alloc((size_t)B * 256 * 3 * 4);
    float*    l3x   = (float*)alloc((size_t)B * 64 * 3 * 4);
    float*    l1p   = (float*)alloc((size_t)B * 1024 * 64 * 4);
    float*    l2p   = (float*)alloc((size_t)B * 256 * 128 * 4);
    float*    l3p   = (float*)alloc((size_t)B * 64 * 256 * 4);
    float*    l4p   = (float*)alloc((size_t)B * 1024 * 4);
    float*    f3    = (float*)alloc((size_t)B * 64 * 256 * 4);
    float*    f2    = (float*)alloc((size_t)B * 256 * 256 * 4);
    float*    f1    = (float*)alloc((size_t)B * 1024 * 128 * 4);

    // One linear (+optional BN/ReLU) layer: fbuf(M,Ci) -> fbuf(M,Co) (or bnOut)
    auto run_layer = [&](int li, long M, bool bn, float* bnOut) {
        const Layer& L = Ls[li];
        int Kp = (L.Ci + 31) & ~31;
        int ksteps = Kp >> 5;
        int ntiles = (L.Co + 15) >> 4;
        int ntilesP = (ntiles + 3) & ~3;           // multiple of 4 for 4-tile waves
        long tA = M * Kp;
        cvt_pad_f16_kernel<<<gblocks(tA), 256, 0, stream>>>(fbuf, a16, M, L.Ci, Kp);
        long tW = (long)ntilesP * ksteps * 512;
        pack_weights_kernel<<<gblocks(tW), 256, 0, stream>>>(L.W, wpk, L.Co, L.Ci, Kp, tW);
        dim3 gg((unsigned)(M / 64), (unsigned)(ntilesP / 4));
        gemm_wmma_kernel<<<gg, 128, 0, stream>>>(a16, wpk, L.bias, fbuf, (int)M, Kp, L.Co);
        if (bn) {
            bn_stats_kernel<<<L.Co, 256, 0, stream>>>(fbuf, stats, stats + 1024, M, L.Co);
            long tE = M * L.Co;
            bn_relu_kernel<<<gblocks(tE), 256, 0, stream>>>(
                fbuf, bnOut ? bnOut : fbuf, stats, stats + 1024, tE, L.Co);
        }
    };

    auto run_sa = [&](const float* xyz, int Np, const float* pts, int Cp,
                      int npoint, float r2, int firstLayer, int nLayers,
                      float* new_xyz, float* outP) {
        fps_kernel<<<B, 256, 0, stream>>>(xyz, fpsb, Np, npoint);
        long q = (long)B * npoint;
        gather_new_xyz_kernel<<<gblocks(q), 256, 0, stream>>>(xyz, fpsb, new_xyz, Np, npoint, q);
        ball_query_kernel<<<gblocks(q), 256, 0, stream>>>(xyz, new_xyz, idxb, Np, npoint, r2, 32, q);
        long M = q * 32;
        long tG = M * (3 + Cp);
        group_points_kernel<<<gblocks(tG), 256, 0, stream>>>(
            xyz, new_xyz, pts, idxb, fbuf, Np, npoint, 32, Cp, tG);
        for (int l = 0; l < nLayers; ++l) run_layer(firstLayer + l, M, true, nullptr);
        int Cout = Ls[firstLayer + nLayers - 1].Co;
        long tP = q * Cout;
        maxpool_kernel<<<gblocks(tP), 256, 0, stream>>>(fbuf, outP, q, 32, Cout, tP);
    };

    auto run_fp = [&](const float* xyz1, int Nq, const float* xyz2, int S,
                      const float* pts1, int C1, const float* pts2, int C2,
                      int firstLayer, int nLayers, float* outP) {
        long q = (long)B * Nq;
        if (S > 1)
            knn3_kernel<<<gblocks(q), 256, 0, stream>>>(xyz1, xyz2, i3b, wtb, Nq, S, q);
        long t = q * (C1 + C2);
        interp_concat_kernel<<<gblocks(t), 256, 0, stream>>>(
            pts1, pts2, i3b, wtb, fbuf, t, C1, C2, S, Nq);
        for (int l = 0; l < nLayers; ++l)
            run_layer(firstLayer + l, q, true, (l == nLayers - 1) ? outP : nullptr);
    };

    // ---- pipeline ----
    long tT = (long)B * N * 3;
    transpose_in_kernel<<<gblocks(tT), 256, 0, stream>>>(x, xyzT, B, 3, N, tT);

    run_sa(xyzT, N,    nullptr, 0,   1024, 0.01f, 0, 3, l1x, l1p);
    run_sa(l1x,  1024, l1p,     64,  256,  0.04f, 3, 3, l2x, l2p);
    run_sa(l2x,  256,  l2p,     128, 64,   0.16f, 6, 3, l3x, l3p);

    { // sa4 (group-all)
        long bs = (long)B * 64;
        long t = bs * 259;
        concat2_kernel<<<gblocks(t), 256, 0, stream>>>(l3x, l3p, fbuf, bs, 3, 256, t);
        for (int l = 0; l < 3; ++l) run_layer(9 + l, bs, true, nullptr);
        long tP = (long)B * 1024;
        maxpool_kernel<<<gblocks(tP), 256, 0, stream>>>(fbuf, l4p, (long)B, 64, 1024, tP);
    }

    run_fp(l3x,  64,   nullptr, 1,    l3p,  256, l4p, 1024, 12, 2, f3);
    run_fp(l2x,  256,  l3x,     64,   l2p,  128, f3,  256,  14, 2, f2);
    run_fp(l1x,  1024, l2x,     256,  l1p,  64,  f2,  256,  16, 2, f1);
    run_fp(xyzT, 4096, l1x,     1024, xyzT, 3,   f1,  128,  18, 3, nullptr);

    { // head
        long M = (long)B * N;
        run_layer(21, M, true, nullptr);
        run_layer(22, M, false, nullptr);  // final linear, bias, no BN
        long t = (long)B * 13 * N;
        transpose_out_kernel<<<gblocks(t), 256, 0, stream>>>(fbuf, (float*)d_out, B, N, 13, t);
    }
}
